// MossFormer_56126632624632
// MI455X (gfx1250) — matrix operand order repaired and checked by reading, compiled, not run on hardware
//
#include <hip/hip_runtime.h>
#include <hip/hip_bf16.h>
#include <math.h>

#define G_    64
#define S_    256
#define DIM_  512
#define HID_  2048
#define QK_   128
#define KW_   17
#define GS_   (G_*S_)

typedef __attribute__((ext_vector_type(16))) __bf16          v16bf;
typedef __attribute__((ext_vector_type(8)))  float           v8f;
typedef __attribute__((ext_vector_type(8)))  unsigned short  ushort8;

union Frag16 { unsigned short u[16]; ushort8 h[2]; v16bf v; };

__device__ __forceinline__ unsigned short f32_to_bf16(float f) {
  unsigned int u = __builtin_bit_cast(unsigned int, f);
  unsigned int r = u + 0x7FFFu + ((u >> 16) & 1u);
  return (unsigned short)(r >> 16);
}
__device__ __forceinline__ unsigned short to_bf16(float v) { return f32_to_bf16(v); }
__device__ __forceinline__ unsigned short to_bf16(unsigned short v) { return v; }
__device__ __forceinline__ float sigmoidf_(float x) { return 1.0f / (1.0f + expf(-x)); }

__device__ float block_reduce_sum(float v, float* smem) {
  __syncthreads();
  #pragma unroll
  for (int o = 16; o > 0; o >>= 1) v += __shfl_down(v, o, 32);
  int wid = threadIdx.x >> 5, ln = threadIdx.x & 31;
  if (ln == 0) smem[wid] = v;
  __syncthreads();
  int nw = blockDim.x >> 5;
  v = (threadIdx.x < (unsigned)nw) ? smem[threadIdx.x] : 0.0f;
  if (wid == 0) {
    #pragma unroll
    for (int o = 4; o > 0; o >>= 1) v += __shfl_down(v, o, 32);
    if (ln == 0) smem[0] = v;
  }
  __syncthreads();
  float r = smem[0];
  __syncthreads();
  return r;
}

// ---------------------------------------------------------------------------
// Batched WMMA GEMM: D[b] = epilogue( A[b] (MxK, K contiguous) *
//                                     B[b] (KxN, supplied K-contiguous per n) )
// bf16 in, f32 accumulate via v_wmma_f32_16x16x32_bf16. One wave per 16x16
// tile; all fragment loads are 16B vector loads. Output strides (ors/ocs)
// allow storing a transposed result directly.
// ---------------------------------------------------------------------------
enum Ep { EP_STORE = 0, EP_ADD = 1, EP_BIAS_SILU = 2, EP_RELUSQ = 3 };

template <int EPI, bool OUTBF, bool ZDIAG>
__global__ __launch_bounds__(256) void wmma_gemm_kernel(
    const unsigned short* __restrict__ A, const unsigned short* __restrict__ B,
    void* __restrict__ Out, const float* __restrict__ bias,
    int M, int N, int Kd,
    long long asb, long long ars,
    long long bsb, long long bcs,
    long long osb, long long ors, long long ocs, float scale)
{
  const int wave = threadIdx.x >> 5;
  const int lane = threadIdx.x & 31;
  const int tiles_m = M >> 4, tiles_n = N >> 4;
  const long long tile = (long long)blockIdx.y * 8 + wave;
  if (tile >= (long long)tiles_m * (long long)tiles_n) return;  // wave-uniform, EXEC stays full
  const int tm = (int)(tile % tiles_m);
  const int tn = (int)(tile / tiles_m);
  const unsigned short* Ab = A + (long long)blockIdx.x * asb;
  const unsigned short* Bb = B + (long long)blockIdx.x * bsb;
  const int l15 = lane & 15;
  const bool hi = lane >= 16;
  const unsigned short* pa0 = Ab + (long long)(tm * 16 + l15) * ars + (hi ? 8 : 0);
  const unsigned short* pb0 = Bb + (long long)(tn * 16 + l15) * bcs + (hi ? 8 : 0);

  v8f acc = {};
  for (int k0 = 0; k0 < Kd; k0 += 32) {
    Frag16 af, bfr;
    // A fragment: lane<16 holds K = ka..ka+7 and ka+16..ka+23 (ka = k0 or k0+8)
    af.h[0] = *(const ushort8*)(pa0 + k0);
    af.h[1] = *(const ushort8*)(pa0 + k0 + 16);
    // B fragment: lane<16 holds K = kb..kb+7 and kb+16..kb+23 at one column (kb = k0 or k0+8)
    bfr.h[0] = *(const ushort8*)(pb0 + k0);
    bfr.h[1] = *(const ushort8*)(pb0 + k0 + 16);
    acc = __builtin_amdgcn_wmma_f32_16x16x32_bf16(false, af.v, false, bfr.v,
                                                  (short)0, acc, false, false);
  }

  const int nIdx = tn * 16 + l15;
  const int mBase = tm * 16 + (hi ? 8 : 0);
  float bval = (EPI == EP_BIAS_SILU) ? bias[nIdx] : 0.0f;
  #pragma unroll
  for (int i = 0; i < 8; ++i) {
    int m = mBase + i;
    float v = acc[i];
    if (EPI == EP_BIAS_SILU) { v += bval; v = v * sigmoidf_(v); }
    else if (EPI == EP_RELUSQ) {
      v *= scale; v = v > 0.f ? v : 0.f; v = v * v;
      if (ZDIAG && m == nIdx) v = 0.f;
    } else { v *= scale; }
    long long o = (long long)blockIdx.x * osb + (long long)m * ors + (long long)nIdx * ocs;
    if (OUTBF) ((unsigned short*)Out)[o] = f32_to_bf16(v);
    else { float* O = (float*)Out; if (EPI == EP_ADD) O[o] += v; else O[o] = v; }
  }
}

template <int EPI, bool OUTBF, bool ZDIAG>
static void run_gemm(hipStream_t st, const void* A, const void* B, void* Out, const float* bias,
                     int M, int N, int K, int batch,
                     long long asb, long long ars,
                     long long bsb, long long bcs,
                     long long osb, long long ors, long long ocs, float scale) {
  long long tiles = (long long)(M / 16) * (long long)(N / 16);
  dim3 grid((unsigned)batch, (unsigned)((tiles + 7) / 8));
  wmma_gemm_kernel<EPI, OUTBF, ZDIAG><<<grid, 256, 0, st>>>(
      (const unsigned short*)A, (const unsigned short*)B, Out, bias,
      M, N, K, asb, ars, bsb, bcs, osb, ors, ocs, scale);
}

// ---------------------------------------------------------------------------
// LDS-tiled transpose (+ optional f32->bf16 cast fused in).
// out[b][r][c] = bf16( in[b*ibs + r*irs + c*ics] ), out contiguous in c.
// 32x32 tiles, 256 threads, padded LDS -> conflict-free, coalesced both sides
// (call with irs == 1 so reads are contiguous along tx).
// ---------------------------------------------------------------------------
template <typename T>
__global__ __launch_bounds__(256) void transpose_bf16_kernel(
    const T* __restrict__ in, unsigned short* __restrict__ out,
    int R, int C, long long ibs, long long irs, long long ics)
{
  __shared__ unsigned short tile[32][33];
  int b = blockIdx.z;
  int r0 = blockIdx.y << 5, c0 = blockIdx.x << 5;
  int tx = threadIdx.x & 31, ty = threadIdx.x >> 5;   // 32 x 8
  long long base = (long long)b * ibs;
  #pragma unroll
  for (int i = 0; i < 4; ++i) {
    int c = c0 + ty + i * 8;
    tile[ty + i * 8][tx] = to_bf16(in[base + (long long)(r0 + tx) * irs + (long long)c * ics]);
  }
  __syncthreads();
  #pragma unroll
  for (int i = 0; i < 4; ++i) {
    int r = r0 + ty + i * 8;
    out[((long long)b * R + r) * C + (c0 + tx)] = tile[tx][ty + i * 8];
  }
}

// ---------------------------------------------------------------------------
// Elementwise / LN / conv kernels
// ---------------------------------------------------------------------------

// Fold LN gamma into weights, store transposed (N,K) bf16 for contiguous B frags.
__global__ void prep_wt_kernel(const float* __restrict__ W, const float* __restrict__ g,
                               unsigned short* __restrict__ Wt, int D, int N, long long total) {
  long long idx = (long long)blockIdx.x * blockDim.x + threadIdx.x;
  if (idx >= total) return;
  int d = (int)(idx / N), n = (int)(idx % N);
  Wt[(long long)n * D + d] = f32_to_bf16(g[d] * W[idx]);
}

// bias'[n] = bias[n] + sum_d ln_b[d] * W[d][n]
__global__ __launch_bounds__(256) void prep_bias_kernel(const float* __restrict__ W,
                                                        const float* __restrict__ lnb,
                                                        const float* __restrict__ bias,
                                                        float* __restrict__ out, int D, int N) {
  __shared__ float smem[8];
  int n = blockIdx.x;
  float s = 0.f;
  for (int d = threadIdx.x; d < D; d += 256) s += lnb[d] * W[(long long)d * N + n];
  s = block_reduce_sum(s, smem);
  if (threadIdx.x == 0) out[n] = bias[n] + s;
}

// LN over the shifted-concat input nx (gamma/beta folded into weights).
__global__ __launch_bounds__(128) void ln1_kernel(const float* __restrict__ x,
                                                  unsigned short* __restrict__ out) {
  __shared__ float smem[8];
  int rowIdx = blockIdx.x;            // g*S + s
  int s = rowIdx & (S_ - 1);
  long long rbase = (long long)rowIdx * DIM_;
  long long pbase = rbase - DIM_;     // previous sequence row
  float vals[4]; float sum = 0.f;
  #pragma unroll
  for (int i = 0; i < 4; ++i) {
    int d = threadIdx.x + i * 128;
    float v = (d < 256) ? ((s > 0) ? x[pbase + d] : 0.f) : x[rbase + d];
    vals[i] = v; sum += v;
  }
  float mean = block_reduce_sum(sum, smem) * (1.0f / DIM_);
  float var = 0.f;
  #pragma unroll
  for (int i = 0; i < 4; ++i) { float d = vals[i] - mean; var += d * d; }
  var = block_reduce_sum(var, smem) * (1.0f / DIM_);
  float r = rsqrtf(var + 1e-5f);
  #pragma unroll
  for (int i = 0; i < 4; ++i)
    out[rbase + threadIdx.x + i * 128] = f32_to_bf16((vals[i] - mean) * r);
}

// LN over 1024-wide gated output.
__global__ __launch_bounds__(256) void ln2_kernel(const float* __restrict__ in,
                                                  unsigned short* __restrict__ out) {
  __shared__ float smem[8];
  long long rbase = (long long)blockIdx.x * 1024;
  float vals[4]; float sum = 0.f;
  #pragma unroll
  for (int i = 0; i < 4; ++i) { float v = in[rbase + threadIdx.x + i * 256]; vals[i] = v; sum += v; }
  float mean = block_reduce_sum(sum, smem) * (1.0f / 1024.0f);
  float var = 0.f;
  #pragma unroll
  for (int i = 0; i < 4; ++i) { float d = vals[i] - mean; var += d * d; }
  var = block_reduce_sum(var, smem) * (1.0f / 1024.0f);
  float r = rsqrtf(var + 1e-5f);
  #pragma unroll
  for (int i = 0; i < 4; ++i)
    out[rbase + threadIdx.x + i * 256] = f32_to_bf16((vals[i] - mean) * r);
}

// out = (resid?) + in + depthwise_conv17(in); layout (G,S,C)
__global__ void dwconv_kernel(const float* __restrict__ in, const float* __restrict__ w,
                              const float* __restrict__ resid, float* __restrict__ out,
                              int C, long long total) {
  long long idx = (long long)blockIdx.x * blockDim.x + threadIdx.x;
  if (idx >= total) return;
  int c = (int)(idx % C);
  int t = (int)((idx / C) % S_);
  float acc = in[idx];
  #pragma unroll
  for (int k = 0; k < KW_; ++k) {
    int tt = t + k - (KW_ / 2);
    if (tt >= 0 && tt < S_) acc += in[idx + (long long)(tt - t) * C] * w[c * KW_ + k];
  }
  out[idx] = resid ? (resid[idx] + acc) : acc;
}

// qq/lq/kk/lk = bf16(qk * gamma[j] + beta[j]), 4 stacked buffers
__global__ void scale4_kernel(const float* __restrict__ qk, const float* __restrict__ gamma,
                              const float* __restrict__ beta, unsigned short* __restrict__ out,
                              long long total) {
  long long idx = (long long)blockIdx.x * blockDim.x + threadIdx.x;
  if (idx >= total) return;
  int d = (int)(idx % QK_);
  float v = qk[idx];
  #pragma unroll
  for (int j = 0; j < 4; ++j)
    out[(long long)j * total + idx] = f32_to_bf16(v * gamma[j * QK_ + d] + beta[j * QK_ + d]);
}

// out = att_u * v * sigmoid(att_v * u); att/h are (GS,2048) with v|u halves
__global__ void gate_kernel(const float* __restrict__ att, const float* __restrict__ h,
                            float* __restrict__ outg, long long total) {
  long long idx = (long long)blockIdx.x * blockDim.x + threadIdx.x;
  if (idx >= total) return;
  long long row = idx >> 10;
  int j = (int)(idx & 1023);
  float av = att[row * 2048 + j],        au = att[row * 2048 + 1024 + j];
  float vv = h[row * 2048 + j],          uu = h[row * 2048 + 1024 + j];
  outg[idx] = au * vv * sigmoidf_(av * uu);
}

// ---------------------------------------------------------------------------
extern "C" void kernel_launch(void* const* d_in, const int* in_sizes, int n_in,
                              void* d_out, int out_size, void* d_ws, size_t ws_size,
                              hipStream_t stream) {
  (void)in_sizes; (void)n_in; (void)out_size; (void)ws_size;
  const float* x        = (const float*)d_in[0];
  const float* ln_h_g   = (const float*)d_in[1];
  const float* ln_h_b   = (const float*)d_in[2];
  const float* W_h      = (const float*)d_in[3];
  const float* b_h      = (const float*)d_in[4];
  const float* conv_h   = (const float*)d_in[5];
  const float* ln_qk_g  = (const float*)d_in[6];
  const float* ln_qk_b  = (const float*)d_in[7];
  const float* W_qk     = (const float*)d_in[8];
  const float* b_qk     = (const float*)d_in[9];
  const float* conv_qk  = (const float*)d_in[10];
  const float* os_gamma = (const float*)d_in[11];
  const float* os_beta  = (const float*)d_in[12];
  const float* ln_o_g   = (const float*)d_in[13];
  const float* ln_o_b   = (const float*)d_in[14];
  const float* W_o      = (const float*)d_in[15];
  const float* b_o      = (const float*)d_in[16];
  const float* conv_o   = (const float*)d_in[17];

  // Workspace carve (256B aligned)
  char* ws = (char*)d_ws; size_t off = 0;
  auto carve = [&](size_t bytes) -> char* {
    char* p = ws + off; off += (bytes + 255) & ~(size_t)255; return p;
  };
  unsigned short* ln1b  = (unsigned short*)carve((size_t)GS_ * DIM_ * 2);
  unsigned short* Whp   = (unsigned short*)carve((size_t)HID_ * DIM_ * 2);
  float*          bhp   = (float*)carve(HID_ * 4);
  unsigned short* Wqkp  = (unsigned short*)carve((size_t)QK_ * DIM_ * 2);
  float*          bqkp  = (float*)carve(QK_ * 4);
  unsigned short* Wop   = (unsigned short*)carve((size_t)DIM_ * 1024 * 2);
  float*          bop   = (float*)carve(DIM_ * 4);
  float*          h_lin = (float*)carve((size_t)GS_ * HID_ * 4);
  float*          h     = (float*)carve((size_t)GS_ * HID_ * 4);
  unsigned short* h_bfT = (unsigned short*)carve((size_t)G_ * HID_ * S_ * 2);   // (G,2048,256)
  unsigned short* crsT  = (unsigned short*)carve((size_t)S_ * HID_ * G_ * 2);   // (S,2048,64)
  float*          qk_l  = (float*)carve((size_t)GS_ * QK_ * 4);
  float*          qk    = (float*)carve((size_t)GS_ * QK_ * 4);
  unsigned short* qlkk  = (unsigned short*)carve((size_t)4 * GS_ * QK_ * 2);
  unsigned short* lkT   = (unsigned short*)carve((size_t)G_ * QK_ * S_ * 2);    // (G,128,256)
  unsigned short* a_loc = (unsigned short*)carve((size_t)G_ * S_ * S_ * 2);
  unsigned short* a_crs = (unsigned short*)carve((size_t)S_ * G_ * G_ * 2);
  unsigned short* kvT   = (unsigned short*)carve((size_t)G_ * HID_ * QK_ * 2);  // (G,2048,128)
  float*          att   = h_lin;  // h_lin dead after conv_h -> reuse as attention accumulator
  float*          out_g = (float*)carve((size_t)GS_ * 1024 * 4);
  unsigned short* ln2b  = (unsigned short*)carve((size_t)GS_ * 1024 * 2);
  float*          o_lin = (float*)carve((size_t)GS_ * DIM_ * 4);

  const int TB = 256;
  auto blocks = [&](long long n) { return dim3((unsigned)((n + TB - 1) / TB)); };

  // --- Fold LN affine into weights (transposed bf16) + folded biases
  prep_wt_kernel<<<blocks((long long)DIM_ * HID_), TB, 0, stream>>>(W_h, ln_h_g, Whp, DIM_, HID_, (long long)DIM_ * HID_);
  prep_bias_kernel<<<HID_, TB, 0, stream>>>(W_h, ln_h_b, b_h, bhp, DIM_, HID_);
  prep_wt_kernel<<<blocks((long long)DIM_ * QK_), TB, 0, stream>>>(W_qk, ln_qk_g, Wqkp, DIM_, QK_, (long long)DIM_ * QK_);
  prep_bias_kernel<<<QK_, TB, 0, stream>>>(W_qk, ln_qk_b, b_qk, bqkp, DIM_, QK_);
  prep_wt_kernel<<<blocks((long long)1024 * DIM_), TB, 0, stream>>>(W_o, ln_o_g, Wop, 1024, DIM_, (long long)1024 * DIM_);
  prep_bias_kernel<<<DIM_, TB, 0, stream>>>(W_o, ln_o_b, b_o, bop, 1024, DIM_);

  // --- Shift-concat + LN -> bf16 activations (shared by h and qk branches)
  ln1_kernel<<<GS_, 128, 0, stream>>>(x, ln1b);

  // --- h_lin = silu(ln @ W_h' + b'), qk_lin = silu(ln @ W_qk' + b')
  run_gemm<EP_BIAS_SILU, false, false>(stream, ln1b, Whp, h_lin, bhp,
      GS_, HID_, DIM_, 1,  0, DIM_,  0, DIM_,  0, HID_, 1, 1.0f);
  run_gemm<EP_BIAS_SILU, false, false>(stream, ln1b, Wqkp, qk_l, bqkp,
      GS_, QK_, DIM_, 1,  0, DIM_,  0, DIM_,  0, QK_, 1, 1.0f);

  // --- depthwise conv (x + conv(x))
  dwconv_kernel<<<blocks((long long)GS_ * HID_), TB, 0, stream>>>(h_lin, conv_h, nullptr, h, HID_, (long long)GS_ * HID_);
  dwconv_kernel<<<blocks((long long)GS_ * QK_), TB, 0, stream>>>(qk_l, conv_qk, nullptr, qk, QK_, (long long)GS_ * QK_);

  // --- transposed bf16 views of [v|u] (cast fused into LDS-tiled transpose)
  {
    dim3 g1(S_ / 32, HID_ / 32, G_);          // h_bfT[g][j][s]
    transpose_bf16_kernel<float><<<g1, 256, 0, stream>>>(h, h_bfT,
        HID_, S_, (long long)S_ * HID_, 1, (long long)HID_);
    dim3 g2(G_ / 32, HID_ / 32, S_);          // crsT[s][j][g]
    transpose_bf16_kernel<float><<<g2, 256, 0, stream>>>(h, crsT,
        HID_, G_, (long long)HID_, 1, (long long)S_ * HID_);
  }

  scale4_kernel<<<blocks((long long)GS_ * QK_), TB, 0, stream>>>(qk, os_gamma, os_beta, qlkk, (long long)GS_ * QK_);

  const unsigned short* qq = qlkk;
  const unsigned short* lq = qlkk + (size_t)1 * GS_ * QK_;
  const unsigned short* kk = qlkk + (size_t)2 * GS_ * QK_;
  const unsigned short* lk = qlkk + (size_t)3 * GS_ * QK_;

  {
    dim3 g3(S_ / 32, QK_ / 32, G_);           // lkT[g][d][t]
    transpose_bf16_kernel<unsigned short><<<g3, 256, 0, stream>>>(lk, lkT,
        QK_, S_, (long long)S_ * QK_, 1, (long long)QK_);
  }

  // --- local scores: relu(qq @ kk^T / 256)^2  (batch = group)
  run_gemm<EP_RELUSQ, true, false>(stream, qq, kk, a_loc, nullptr,
      S_, S_, QK_, G_,
      (long long)S_ * QK_, QK_,
      (long long)S_ * QK_, QK_,
      (long long)S_ * S_, S_, 1, 1.0f / 256.0f);

  // --- cross scores: relu(qq_c @ kk_c^T / G)^2, diag zeroed  (batch = seq pos)
  run_gemm<EP_RELUSQ, true, true>(stream, qq, kk, a_crs, nullptr,
      G_, G_, QK_, S_,
      QK_, (long long)S_ * QK_,
      QK_, (long long)S_ * QK_,
      (long long)G_ * G_, G_, 1, 1.0f / (float)G_);

  // --- quad (local): att = a_loc @ [v|u]
  run_gemm<EP_STORE, false, false>(stream, a_loc, h_bfT, att, nullptr,
      S_, HID_, S_, G_,
      (long long)S_ * S_, S_,
      (long long)HID_ * S_, S_,
      (long long)S_ * HID_, HID_, 1, 1.0f);

  // --- quad (cross): att += a_crs @ [v|u]_c  (batch = seq pos, scattered rows)
  run_gemm<EP_ADD, false, false>(stream, a_crs, crsT, att, nullptr,
      G_, HID_, G_, S_,
      (long long)G_ * G_, G_,
      (long long)HID_ * G_, G_,
      HID_, (long long)S_ * HID_, 1, 1.0f);

  // --- linear attention: kvT = ((lk^T @ [v|u]) / 256)^T stored (G,2048,128) bf16
  run_gemm<EP_STORE, true, false>(stream, lkT, h_bfT, kvT, nullptr,
      QK_, HID_, S_, G_,
      (long long)QK_ * S_, S_,
      (long long)HID_ * S_, S_,
      (long long)HID_ * QK_, 1, QK_, 1.0f / 256.0f);

  // --- att += lq @ kv
  run_gemm<EP_ADD, false, false>(stream, lq, kvT, att, nullptr,
      S_, HID_, QK_, G_,
      (long long)S_ * QK_, QK_,
      (long long)HID_ * QK_, QK_,
      (long long)S_ * HID_, HID_, 1, 1.0f);

  // --- gate: out = att_u * v * sigmoid(att_v * u)
  gate_kernel<<<blocks((long long)GS_ * 1024), TB, 0, stream>>>(att, h, out_g, (long long)GS_ * 1024);

  // --- output FFConvM
  ln2_kernel<<<GS_, 256, 0, stream>>>(out_g, ln2b);
  run_gemm<EP_BIAS_SILU, false, false>(stream, ln2b, Wop, o_lin, bop,
      GS_, DIM_, 1024, 1,  0, 1024,  0, 1024,  0, DIM_, 1, 1.0f);
  dwconv_kernel<<<blocks((long long)GS_ * DIM_), TB, 0, stream>>>(o_lin, conv_o, x, (float*)d_out, DIM_, (long long)GS_ * DIM_);
}
